// Block_2070174236875
// MI455X (gfx1250) — compile-verified, matrix operand order checked by reading, not run on hardware
//
#include <hip/hip_runtime.h>
#include <hip/hip_bf16.h>
#include <math.h>

// ---------------------------------------------------------------------------
// Transformer block on MI455X (gfx1250): all GEMMs + attention through
// v_wmma_f32_16x16x32_bf16 (fp32 data converted to bf16, fp32 accumulate).
// GEMM tiles are staged with gfx1250 async global->LDS loads (ASYNCcnt) and
// double-buffered so WMMA issue overlaps the next tile's DMA.
// ---------------------------------------------------------------------------

typedef __attribute__((ext_vector_type(16))) __bf16 v16bf;
typedef __attribute__((ext_vector_type(8)))  __bf16 v8bf;
typedef __attribute__((ext_vector_type(8)))  float  v8f;

#if defined(__gfx1250__) && __has_builtin(__builtin_amdgcn_global_load_async_to_lds_b128)
#define HAVE_ASYNC_LDS 1
#else
#define HAVE_ASYNC_LDS 0
#endif

#if HAVE_ASYNC_LDS
// Exact parameter types per clang's diagnostic: int __vector(4) pointers,
// global (AS1) source and LDS (AS3) destination.
typedef int v4i_vs __attribute__((vector_size(16)));
typedef __attribute__((address_space(1))) v4i_vs* gas_ptr;
typedef __attribute__((address_space(3))) v4i_vs* las_ptr;
#endif

template <int N>
__device__ __forceinline__ void wait_asynccnt() {
#if HAVE_ASYNC_LDS
#if __has_builtin(__builtin_amdgcn_s_wait_asynccnt)
  __builtin_amdgcn_s_wait_asynccnt(N);
#else
  if constexpr (N == 0) asm volatile("s_wait_asynccnt 0x0" ::: "memory");
  else                  asm volatile("s_wait_asynccnt 0x6" ::: "memory");
#endif
#endif
}

__device__ __forceinline__ v8f vzero8() {
  v8f z;
#pragma unroll
  for (int i = 0; i < 8; ++i) z[i] = 0.0f;
  return z;
}

__device__ __forceinline__ v8f wmma_bf16(v16bf a, v16bf b, v8f c) {
  // (neg_a, A, neg_b, B, c_mod, C, reuse_a, reuse_b)
  return __builtin_amdgcn_wmma_f32_16x16x32_bf16(false, a, false, b, (short)0, c,
                                                 false, false);
}

// Load one 16x32 bf16 A-fragment (or 32x16 B-fragment, by symmetry) whose
// contiguous dimension is K. ISA layout (CDNA5 ISA): lanes 0-15 take K pairs
// {0..7} and {16..23}; lanes 16-31 take {8..15} and {24..31}. So each lane
// reads two contiguous 16-byte chunks at +hi8 and +16+hi8.
__device__ __forceinline__ v16bf load_frag16(const __bf16* base, int lane) {
  const int hi8 = (lane & 16) ? 8 : 0;
  const v8bf lo = *(const v8bf*)(base + hi8);
  const v8bf hi = *(const v8bf*)(base + 16 + hi8);
  v16bf f;
#pragma unroll
  for (int i = 0; i < 8; ++i) { f[i] = lo[i]; f[i + 8] = hi[i]; }
  return f;
}

// ---------------------------------------------------------------------------
// Weight convert + transpose: fp32 [K][N] -> bf16 [N][K] (so GEMM B-fragments
// are contiguous along K).
// ---------------------------------------------------------------------------
__global__ __launch_bounds__(256) void convert_wT_kernel(
    const float* __restrict__ in, __bf16* __restrict__ out, int K, int N) {
  const size_t i = (size_t)blockIdx.x * 256 + threadIdx.x;
  const size_t total = (size_t)K * N;
  if (i >= total) return;
  const int n = (int)(i % N);
  const int k = (int)(i / N);
  out[(size_t)n * K + k] = (__bf16)in[i];
}

// ---------------------------------------------------------------------------
// LayerNorm (row length 768) -> bf16. One block (256 threads) per row.
// ---------------------------------------------------------------------------
__global__ __launch_bounds__(256) void ln_bf16_kernel(
    const float* __restrict__ x, const float* __restrict__ g,
    const float* __restrict__ b, __bf16* __restrict__ out) {
  const int row = blockIdx.x;
  const int tid = threadIdx.x;
  const float* xr = x + (size_t)row * 768;
  float v0 = xr[tid], v1 = xr[tid + 256], v2 = xr[tid + 512];

  __shared__ float sm[8];
  float s = v0 + v1 + v2;
#pragma unroll
  for (int o = 16; o > 0; o >>= 1) s += __shfl_xor(s, o, 32);
  if ((tid & 31) == 0) sm[tid >> 5] = s;
  __syncthreads();
  float tot = 0.0f;
#pragma unroll
  for (int i = 0; i < 8; ++i) tot += sm[i];
  const float mu = tot * (1.0f / 768.0f);
  __syncthreads();

  float d0 = v0 - mu, d1 = v1 - mu, d2 = v2 - mu;
  float s2 = d0 * d0 + d1 * d1 + d2 * d2;
#pragma unroll
  for (int o = 16; o > 0; o >>= 1) s2 += __shfl_xor(s2, o, 32);
  if ((tid & 31) == 0) sm[tid >> 5] = s2;
  __syncthreads();
  float tot2 = 0.0f;
#pragma unroll
  for (int i = 0; i < 8; ++i) tot2 += sm[i];
  const float rs = rsqrtf(tot2 * (1.0f / 768.0f) + 1e-6f);

  __bf16* o_r = out + (size_t)row * 768;
  o_r[tid]       = (__bf16)(d0 * rs * g[tid]       + b[tid]);
  o_r[tid + 256] = (__bf16)(d1 * rs * g[tid + 256] + b[tid + 256]);
  o_r[tid + 512] = (__bf16)(d2 * rs * g[tid + 512] + b[tid + 512]);
}

// ---------------------------------------------------------------------------
// WMMA GEMM: C[M,N] = A[M,K](bf16,row-major) * Bt[N,K](bf16,"KxN" transposed)
// Block tile 128x64, BK=64, 256 threads = 8 waves arranged 4(M)x2(N),
// each wave computes a 32x32 tile = 2x2 WMMA accumulators.
// Tiles are double-buffered in LDS and staged with async global->LDS B128
// loads (6 per thread per stage: 4 for A, 2 for B); s_wait_asynccnt 6 keeps
// one stage in flight while WMMAs run on the other buffer.
// Epilogues: 0 = +bias -> bf16 ; 1 = +bias +residual -> fp32 ;
//            2 = +bias, exact GELU -> bf16.
// Dims must be multiples of the tile (true here: M=8192, N in {768,2304,3072},
// K in {768,3072}).
// ---------------------------------------------------------------------------
#define G_LDK 80  // padded LDS K-stride (bf16 elems): 160B rows keep 16B align

template <int EPI>
__global__ __launch_bounds__(256, 2) void gemm_wmma_kernel(
    const __bf16* __restrict__ A, const __bf16* __restrict__ Bt,
    const float* __restrict__ bias, const float* __restrict__ resid,
    void* __restrict__ out, int M, int N, int K) {
  __shared__ __bf16 As[2][128 * G_LDK];
  __shared__ __bf16 Bs[2][64 * G_LDK];

  const int tid = threadIdx.x;
  const int wid = tid >> 5;
  const int lane = tid & 31;
  const int lane15 = lane & 15;
  const int rhi = (lane & 16) ? 8 : 0;
  const int waveM = wid & 3;   // 4 waves along M (32 rows each)
  const int waveN = wid >> 2;  // 2 waves along N (32 cols each)
  const int tileM = blockIdx.y * 128;
  const int tileN = blockIdx.x * 64;

  // Per-thread staging coordinates (A: 4 chunks of 16B, B: 2 chunks of 16B).
  const int arow0 = tid >> 3;            // + i*32 for i in 0..3
  const int akp = (tid & 7) * 8;
  const int brow0 = tid >> 3;            // + i*32 for i in 0..1
  const int bkp = akp;

  auto issue_stage = [&](int buf, int k0) {
#if HAVE_ASYNC_LDS
#pragma unroll
    for (int i = 0; i < 4; ++i) {
      const int row = arow0 + i * 32;
      __builtin_amdgcn_global_load_async_to_lds_b128(
          (gas_ptr)(A + (size_t)(tileM + row) * K + k0 + akp),
          (las_ptr)(&As[buf][row * G_LDK + akp]), 0, 0);
    }
#pragma unroll
    for (int i = 0; i < 2; ++i) {
      const int row = brow0 + i * 32;
      __builtin_amdgcn_global_load_async_to_lds_b128(
          (gas_ptr)(Bt + (size_t)(tileN + row) * K + k0 + bkp),
          (las_ptr)(&Bs[buf][row * G_LDK + bkp]), 0, 0);
    }
#else
#pragma unroll
    for (int i = 0; i < 4; ++i) {
      const int row = arow0 + i * 32;
      *(v8bf*)(&As[buf][row * G_LDK + akp]) =
          *(const v8bf*)(A + (size_t)(tileM + row) * K + k0 + akp);
    }
#pragma unroll
    for (int i = 0; i < 2; ++i) {
      const int row = brow0 + i * 32;
      *(v8bf*)(&Bs[buf][row * G_LDK + bkp]) =
          *(const v8bf*)(Bt + (size_t)(tileN + row) * K + k0 + bkp);
    }
#endif
  };

  v8f acc[2][2];
#pragma unroll
  for (int mi = 0; mi < 2; ++mi)
#pragma unroll
    for (int ni = 0; ni < 2; ++ni) acc[mi][ni] = vzero8();

  const int nstage = K >> 6;  // K / 64
  issue_stage(0, 0);
  for (int s = 0; s < nstage; ++s) {
    const int cur = s & 1;
    if (s + 1 < nstage) {
      issue_stage(cur ^ 1, (s + 1) * 64);
      wait_asynccnt<6>();   // stage s complete; stage s+1 still in flight
    } else {
      wait_asynccnt<0>();
    }
    __syncthreads();        // stage s visible to all waves

    const __bf16* as = As[cur];
    const __bf16* bs = Bs[cur];
#pragma unroll
    for (int kk = 0; kk < 64; kk += 32) {
      v16bf af[2], bf[2];
#pragma unroll
      for (int mi = 0; mi < 2; ++mi)
        af[mi] = load_frag16(&as[(waveM * 32 + mi * 16 + lane15) * G_LDK + kk], lane);
#pragma unroll
      for (int ni = 0; ni < 2; ++ni)
        bf[ni] = load_frag16(&bs[(waveN * 32 + ni * 16 + lane15) * G_LDK + kk], lane);
#pragma unroll
      for (int mi = 0; mi < 2; ++mi)
#pragma unroll
        for (int ni = 0; ni < 2; ++ni)
          acc[mi][ni] = wmma_bf16(af[mi], bf[ni], acc[mi][ni]);
    }
    __syncthreads();        // buffer `cur` free for stage s+2
  }

  // Epilogue. C/D layout: VGPR r <-> M = r (lanes 0-15) / r+8 (lanes 16-31),
  // lane&15 = N.
#pragma unroll
  for (int mi = 0; mi < 2; ++mi)
#pragma unroll
    for (int ni = 0; ni < 2; ++ni) {
      const int col = tileN + waveN * 32 + ni * 16 + lane15;
      const float bv = bias[col];
#pragma unroll
      for (int r = 0; r < 8; ++r) {
        const int row = tileM + waveM * 32 + mi * 16 + rhi + r;
        const size_t idx = (size_t)row * N + col;
        float v = acc[mi][ni][r] + bv;
        if (EPI == 0) {
          ((__bf16*)out)[idx] = (__bf16)v;
        } else if (EPI == 1) {
          ((float*)out)[idx] = v + resid[idx];
        } else {
          const float ge = 0.5f * v * (1.0f + erff(v * 0.70710678118654752f));
          ((__bf16*)out)[idx] = (__bf16)ge;
        }
      }
    }
}

// ---------------------------------------------------------------------------
// Flash attention with WMMA. Grid: (N/64, B*H). Block: 128 threads = 4 waves,
// each wave owns 16 query rows. Keys streamed in chunks of 32, K/V staged in
// LDS shared by the block. Softmax is done online in the WMMA C-layout;
// P is round-tripped through wave-private LDS to re-layout into an A-fragment.
// ---------------------------------------------------------------------------
__global__ __launch_bounds__(128) void attn_wmma_kernel(
    const __bf16* __restrict__ qkv,  // [B*N, 2304] : q | k | v, head-major 64
    const int* __restrict__ mask,    // [B, N]
    __bf16* __restrict__ out) {      // [B*N, 768]
  constexpr int NK = 2048, C3 = 2304, HD = 64;
  const int tid = threadIdx.x, wid = tid >> 5, lane = tid & 31;
  const int lane15 = lane & 15;
  const int rhi = (lane & 16) ? 8 : 0;
  const int bh = blockIdx.y;
  const int b = bh / 12, h = bh % 12;
  const int q0 = blockIdx.x * 64 + wid * 16;
  const size_t rowbase = (size_t)b * NK;

  __shared__ __bf16 Ks[32 * 80];      // [key][d]  (row = 160B, 16B aligned)
  __shared__ __bf16 Vt[64 * 40];      // [d][key]  (row = 80B)
  __shared__ __bf16 Pt[4][16 * 40];   // per-wave P tile [qrow][key]

  // Q fragments (16 rows x 64 d) straight from global.
  v16bf qa[2];
  {
    const __bf16* qptr = qkv + (rowbase + q0 + lane15) * C3 + h * HD;
    qa[0] = load_frag16(qptr, lane);
    qa[1] = load_frag16(qptr + 32, lane);
  }

  v8f o[4];
#pragma unroll
  for (int ni = 0; ni < 4; ++ni) o[ni] = vzero8();
  float mrun[8], lrun[8];
#pragma unroll
  for (int r = 0; r < 8; ++r) { mrun[r] = -3.0e38f; lrun[r] = 0.0f; }

  for (int kb = 0; kb < NK; kb += 32) {
    __syncthreads();
    {  // Stage K (row-major) and V (transposed) chunks: 128 threads.
      const int key = tid >> 2;        // 0..31
      const int dp = (tid & 3) * 16;   // 0,16,32,48
      const __bf16* kg = qkv + (rowbase + kb + key) * C3 + 768 + h * HD + dp;
      *(v8bf*)(&Ks[key * 80 + dp]) = *(const v8bf*)kg;
      *(v8bf*)(&Ks[key * 80 + dp + 8]) = *(const v8bf*)(kg + 8);
      const __bf16* vg = qkv + (rowbase + kb + key) * C3 + 1536 + h * HD + dp;
      const v8bf va = *(const v8bf*)vg;
      const v8bf vb = *(const v8bf*)(vg + 8);
#pragma unroll
      for (int j = 0; j < 8; ++j) {
        Vt[(dp + j) * 40 + key] = va[j];
        Vt[(dp + 8 + j) * 40 + key] = vb[j];
      }
    }
    __syncthreads();

    // S = Q * K^T for two 16-key subtiles (inner dim d=64 -> 2 wmma each).
    v8f s0 = vzero8(), s1 = vzero8();
#pragma unroll
    for (int kk = 0; kk < 2; ++kk) {
      const v16bf kf0 = load_frag16(&Ks[lane15 * 80 + kk * 32], lane);
      const v16bf kf1 = load_frag16(&Ks[(16 + lane15) * 80 + kk * 32], lane);
      s0 = wmma_bf16(qa[kk], kf0, s0);
      s1 = wmma_bf16(qa[kk], kf1, s1);
    }

    const float scale = 0.125f;  // D^-0.5, D=64
    const bool ok0 = mask[rowbase + kb + lane15] != 0;
    const bool ok1 = mask[rowbase + kb + 16 + lane15] != 0;
    float ml[8];
#pragma unroll
    for (int r = 0; r < 8; ++r) {
      const float a0 = ok0 ? s0[r] * scale : -3.0e38f;
      const float a1 = ok1 ? s1[r] * scale : -3.0e38f;
      s0[r] = a0; s1[r] = a1;
      ml[r] = fmaxf(a0, a1);
    }
    // Row max across the 16 lanes sharing a row (xor 1,2,4,8 stays in-half).
#pragma unroll
    for (int off = 1; off < 16; off <<= 1)
#pragma unroll
      for (int r = 0; r < 8; ++r) ml[r] = fmaxf(ml[r], __shfl_xor(ml[r], off, 32));

    float alpha[8], pl[8];
#pragma unroll
    for (int r = 0; r < 8; ++r) {
      const float mn = fmaxf(mrun[r], ml[r]);
      alpha[r] = __expf(mrun[r] - mn);
      mrun[r] = mn;
      const float p0 = __expf(s0[r] - mn);
      const float p1 = __expf(s1[r] - mn);
      s0[r] = p0; s1[r] = p1;
      pl[r] = p0 + p1;
    }
#pragma unroll
    for (int off = 1; off < 16; off <<= 1)
#pragma unroll
      for (int r = 0; r < 8; ++r) pl[r] += __shfl_xor(pl[r], off, 32);
#pragma unroll
    for (int r = 0; r < 8; ++r) lrun[r] = lrun[r] * alpha[r] + pl[r];
#pragma unroll
    for (int ni = 0; ni < 4; ++ni)
#pragma unroll
      for (int r = 0; r < 8; ++r) o[ni][r] *= alpha[r];

    // Re-layout P (C-layout) -> A-fragment via wave-private LDS.
    __bf16* pw = Pt[wid];
#pragma unroll
    for (int r = 0; r < 8; ++r) {
      pw[(r + rhi) * 40 + lane15] = (__bf16)s0[r];
      pw[(r + rhi) * 40 + 16 + lane15] = (__bf16)s1[r];
    }
    // Wave-internal DS RAW: DS ops from one wave are in-order; wait them out.
    asm volatile("s_wait_dscnt 0" ::: "memory");
    const v16bf pa = load_frag16(&pw[lane15 * 40], lane);
#pragma unroll
    for (int ni = 0; ni < 4; ++ni) {
      const v16bf vf = load_frag16(&Vt[(ni * 16 + lane15) * 40], lane);
      o[ni] = wmma_bf16(pa, vf, o[ni]);
    }
  }

  // Normalize and store: out[b*N + qrow, h*64 + d] (bf16 -> proj GEMM input).
#pragma unroll
  for (int ni = 0; ni < 4; ++ni)
#pragma unroll
    for (int r = 0; r < 8; ++r) {
      const float v = o[ni][r] / lrun[r];
      const int row = q0 + r + rhi;
      out[(rowbase + row) * 768 + h * HD + ni * 16 + lane15] = (__bf16)v;
    }
}

// ---------------------------------------------------------------------------
// Host-side orchestration.
// ---------------------------------------------------------------------------
extern "C" void kernel_launch(void* const* d_in, const int* in_sizes, int n_in,
                              void* d_out, int out_size, void* d_ws, size_t ws_size,
                              hipStream_t stream) {
  (void)in_sizes; (void)n_in; (void)out_size; (void)ws_size;
  const float* x      = (const float*)d_in[0];
  const int*   mask   = (const int*)d_in[1];
  const float* ln1_g  = (const float*)d_in[2];
  const float* ln1_b  = (const float*)d_in[3];
  const float* qkv_w  = (const float*)d_in[4];
  const float* qkv_b  = (const float*)d_in[5];
  const float* proj_w = (const float*)d_in[6];
  const float* proj_b = (const float*)d_in[7];
  const float* ln2_g  = (const float*)d_in[8];
  const float* ln2_b  = (const float*)d_in[9];
  const float* fc1_w  = (const float*)d_in[10];
  const float* fc1_b  = (const float*)d_in[11];
  const float* fc2_w  = (const float*)d_in[12];
  const float* fc2_b  = (const float*)d_in[13];

  constexpr int ROWS = 8192;   // B*N = 4*2048
  constexpr int C = 768, C3 = 2304, HID = 3072;

  // Workspace layout (bytes, 16B-aligned offsets). ~102 MB total.
  constexpr size_t OFF_WQKV = 0;                               // 768x2304 bf16^T
  constexpr size_t OFF_WPROJ = OFF_WQKV + (size_t)C * C3 * 2;  // 768x768
  constexpr size_t OFF_WFC1 = OFF_WPROJ + (size_t)C * C * 2;   // 768x3072
  constexpr size_t OFF_WFC2 = OFF_WFC1 + (size_t)C * HID * 2;  // 3072x768
  constexpr size_t OFF_H    = OFF_WFC2 + (size_t)HID * C * 2;  // ln out (bf16)
  constexpr size_t OFF_QKV  = OFF_H + (size_t)ROWS * C * 2;    // [ROWS,2304] bf16
  constexpr size_t OFF_ATTN = OFF_QKV + (size_t)ROWS * C3 * 2; // [ROWS,768] bf16
  constexpr size_t OFF_X1   = OFF_ATTN + (size_t)ROWS * C * 2; // [ROWS,768] f32
  constexpr size_t OFF_FC1  = OFF_QKV;  // reuse qkv+attn region after proj

  char* ws = (char*)d_ws;
  __bf16* wqkvT  = (__bf16*)(ws + OFF_WQKV);
  __bf16* wprojT = (__bf16*)(ws + OFF_WPROJ);
  __bf16* wfc1T  = (__bf16*)(ws + OFF_WFC1);
  __bf16* wfc2T  = (__bf16*)(ws + OFF_WFC2);
  __bf16* hbuf   = (__bf16*)(ws + OFF_H);
  __bf16* qkvbuf = (__bf16*)(ws + OFF_QKV);
  __bf16* attnb  = (__bf16*)(ws + OFF_ATTN);
  float*  x1     = (float*)(ws + OFF_X1);
  __bf16* fc1buf = (__bf16*)(ws + OFF_FC1);

  auto cvt = [&](const float* w, __bf16* o, int K, int N) {
    const size_t total = (size_t)K * N;
    convert_wT_kernel<<<(unsigned)((total + 255) / 256), 256, 0, stream>>>(w, o, K, N);
  };
  cvt(qkv_w, wqkvT, C, C3);
  cvt(proj_w, wprojT, C, C);
  cvt(fc1_w, wfc1T, C, HID);
  cvt(fc2_w, wfc2T, HID, C);

  // LN1 -> h (bf16)
  ln_bf16_kernel<<<ROWS, 256, 0, stream>>>(x, ln1_g, ln1_b, hbuf);
  // qkv = h @ qkv_w + qkv_b  (bf16 out)
  gemm_wmma_kernel<0><<<dim3(C3 / 64, ROWS / 128), 256, 0, stream>>>(
      hbuf, wqkvT, qkv_b, nullptr, qkvbuf, ROWS, C3, C);
  // attention (flash, WMMA) -> attnb (bf16, [ROWS, 768])
  attn_wmma_kernel<<<dim3(2048 / 64, 4 * 12), 128, 0, stream>>>(qkvbuf, mask, attnb);
  // x1 = x + attnb @ proj_w + proj_b  (f32)
  gemm_wmma_kernel<1><<<dim3(C / 64, ROWS / 128), 256, 0, stream>>>(
      attnb, wprojT, proj_b, x, x1, ROWS, C, C);
  // LN2 -> h (bf16)
  ln_bf16_kernel<<<ROWS, 256, 0, stream>>>(x1, ln2_g, ln2_b, hbuf);
  // fc1 = gelu(h @ fc1_w + fc1_b)  (bf16)
  gemm_wmma_kernel<2><<<dim3(HID / 64, ROWS / 128), 256, 0, stream>>>(
      hbuf, wfc1T, fc1_b, nullptr, fc1buf, ROWS, HID, C);
  // out = x1 + fc1 @ fc2_w + fc2_b  (f32)
  gemm_wmma_kernel<1><<<dim3(C / 64, ROWS / 128), 256, 0, stream>>>(
      fc1buf, wfc2T, fc2_b, x1, (float*)d_out, ROWS, C, HID);
}
